// BitFeedForward2_19439021982249
// MI455X (gfx1250) — 8-bit matrix operand order repaired and checked by reading, compiled, not run on hardware
//
#include <hip/hip_runtime.h>
#include <hip/hip_bf16.h>

typedef __attribute__((ext_vector_type(8))) int v8i;

__device__ __forceinline__ void async_copy16(unsigned lds_off, const void* g) {
  asm volatile("global_load_async_to_lds_b128 %0, %1, off"
               :
               : "v"(lds_off), "v"(g)
               : "memory");
}

#if __has_builtin(__builtin_amdgcn_s_wait_asynccnt)
#define WAIT_ASYNC0() __builtin_amdgcn_s_wait_asynccnt(0)
#else
#define WAIT_ASYNC0() asm volatile("s_wait_asynccnt 0x0" ::: "memory")
#endif

__global__ __launch_bounds__(32) void init_kernel(float* wsum) {
  if (threadIdx.x < 2) wsum[threadIdx.x] = 0.0f;
}

__global__ __launch_bounds__(256) void abssum_kernel(const float* __restrict__ w,
                                                     float* __restrict__ accum,
                                                     size_t n) {
  __shared__ float red[256];
  size_t i = (size_t)blockIdx.x * blockDim.x + threadIdx.x;
  const size_t stride = (size_t)gridDim.x * blockDim.x;
  float s = 0.0f;
  for (; i < n; i += stride) s += fabsf(w[i]);
  red[threadIdx.x] = s;
  __syncthreads();
  for (int st = 128; st > 0; st >>= 1) {
    if (threadIdx.x < (unsigned)st) red[threadIdx.x] += red[threadIdx.x + st];
    __syncthreads();
  }
  if (threadIdx.x == 0) atomicAdd(accum, red[0]);
}

__global__ __launch_bounds__(256) void wquant_kernel(const float* __restrict__ w,
                                                     signed char* __restrict__ q,
                                                     const float* __restrict__ accum,
                                                     float inv_n, size_t n) {
  const float scale  = fmaxf(accum[0] * inv_n, 1e-5f);
  const float iscale = 1.0f / scale;
  size_t i = (size_t)blockIdx.x * blockDim.x + threadIdx.x;
  const size_t stride = (size_t)gridDim.x * blockDim.x;
  for (; i < n; i += stride) {
    float v = rintf(w[i] * iscale);
    v = fminf(fmaxf(v, -1.0f), 1.0f);
    q[i] = (signed char)(int)v;
  }
}

template <int N>
__global__ __launch_bounds__(256) void fwht_quant_kernel(const float* __restrict__ x,
                                                         signed char* __restrict__ q,
                                                         float* __restrict__ rs) {
  __shared__ float buf[N];
  __shared__ float red[256];
  const int row = blockIdx.x;
  const int tid = threadIdx.x;

  const float* xr = x + (size_t)row * N;
  for (int i = tid; i < N; i += 256) buf[i] = xr[i];
  __syncthreads();

  for (int h = 1; h < N; h <<= 1) {
    for (int idx = tid; idx < (N >> 1); idx += 256) {
      const int i = ((idx & ~(h - 1)) << 1) | (idx & (h - 1));
      const int j = i + h;
      const float a = buf[i];
      const float b = buf[j];
      buf[i] = a + b;
      buf[j] = a - b;
    }
    __syncthreads();
  }

  const float norm = rsqrtf((float)N);
  float mx = 0.0f;
  for (int i = tid; i < N; i += 256) {
    const float v = buf[i] * norm;
    buf[i] = v;
    mx = fmaxf(mx, fabsf(v));
  }
  red[tid] = mx;
  __syncthreads();
  for (int st = 128; st > 0; st >>= 1) {
    if (tid < st) red[tid] = fmaxf(red[tid], red[tid + st]);
    __syncthreads();
  }
  const float maxc  = fmaxf(red[0], 1e-5f);
  const float scale = 127.0f / maxc;
  if (tid == 0) rs[row] = maxc * (1.0f / 127.0f);

  signed char* qr = q + (size_t)row * N;
  for (int i = tid; i < N; i += 256) {
    float v = rintf(buf[i] * scale);
    v = fminf(fmaxf(v, -127.0f), 127.0f);
    qr[i] = (signed char)(int)v;
  }
}

__global__ __launch_bounds__(256) void gemm_iu8_kernel(const signed char* __restrict__ Xq,
                                                       const signed char* __restrict__ Wq,
                                                       const float* __restrict__ rs,
                                                       const float* __restrict__ wsum,
                                                       float inv_wn,
                                                       float* __restrict__ out,
                                                       int N, int K, int relu2) {
  __shared__ signed char Atile[2][128 * 64];
  __shared__ signed char Btile[2][64 * 64];

  const int tid  = threadIdx.x;
  const int lane = tid & 31;
  const int wave = tid >> 5;
  const int hlf  = lane >> 4;
  const int l15  = lane & 15;

  const int m0 = blockIdx.y * 128;
  const int n0 = blockIdx.x * 64;

  v8i acc[4];
#pragma unroll
  for (int i = 0; i < 4; ++i) acc[i] = (v8i){0, 0, 0, 0, 0, 0, 0, 0};

  const int ar0 = tid >> 2;
  const int ac0 = (tid & 3) << 4;
  const int ar1 = (tid + 256) >> 2;
  const int br = tid >> 2;
  const int bc = (tid & 3) << 4;

  auto issue_tile = [&](int k0, int bufsel) {
    const unsigned abase = (unsigned)(uintptr_t)&Atile[bufsel][0];
    const unsigned bbase = (unsigned)(uintptr_t)&Btile[bufsel][0];
    async_copy16(abase + (unsigned)(ar0 * 64 + ac0),
                 &Xq[(size_t)(m0 + ar0) * K + k0 + ac0]);
    async_copy16(abase + (unsigned)(ar1 * 64 + ac0),
                 &Xq[(size_t)(m0 + ar1) * K + k0 + ac0]);
    async_copy16(bbase + (unsigned)(br * 64 + bc),
                 &Wq[(size_t)(n0 + br) * K + k0 + bc]);
  };

  const int ksteps = K >> 6;

  issue_tile(0, 0);
  WAIT_ASYNC0();
  __syncthreads();

  for (int ks = 0; ks < ksteps; ++ks) {
    const int cur = ks & 1;

    if (ks + 1 < ksteps) issue_tile((ks + 1) << 6, cur ^ 1);

    v8i a;
    {
      const signed char* arow = &Atile[cur][(wave * 16 + l15) * 64];
#pragma unroll
      for (int j = 0; j < 8; ++j) {
        const int kk = ((j >> 1) << 4) + (hlf << 3) + ((j & 1) << 2);
        a[j] = *(const int*)(arow + kk);
      }
    }

#pragma unroll
    for (int nf = 0; nf < 4; ++nf) {
      v8i b;
      const signed char* brow = &Btile[cur][(nf * 16 + l15) * 64];
#pragma unroll
      for (int j = 0; j < 8; ++j) {
        const int kk = ((j >> 1) << 4) + (hlf << 3) + ((j & 1) << 2);
        b[j] = *(const int*)(brow + kk);
      }
      acc[nf] = __builtin_amdgcn_wmma_i32_16x16x64_iu8(
           true, a,  true, b, acc[nf],
           false,  false);
    }

    if (ks + 1 < ksteps) {
      WAIT_ASYNC0();
      __syncthreads();
    }
  }

  const float sw = fmaxf(wsum[0] * inv_wn, 1e-5f);
#pragma unroll
  for (int nf = 0; nf < 4; ++nf) {
    const int n = n0 + nf * 16 + l15;
#pragma unroll
    for (int r = 0; r < 8; ++r) {
      const int m = m0 + wave * 16 + hlf * 8 + r;
      float v = (float)acc[nf][r] * rs[m] * sw;
      if (relu2) {
        v = fmaxf(v, 0.0f);
        v = v * v;
      }
      out[(size_t)m * N + n] = v;
    }
  }
}

extern "C" void kernel_launch(void* const* d_in, const int* in_sizes, int n_in,
                              void* d_out, int out_size, void* d_ws, size_t ws_size,
                              hipStream_t stream) {
  (void)in_sizes; (void)n_in; (void)out_size; (void)ws_size;

  const float* x   = (const float*)d_in[0];
  const float* wup = (const float*)d_in[1];
  const float* wdn = (const float*)d_in[2];
  float* out = (float*)d_out;

  const int M = 8192;
  const int Hd = 2048;
  const int Id = 4096;

  char* ws = (char*)d_ws;
  size_t off = 0;
  signed char* Xq1 = (signed char*)(ws + off); off += (size_t)M * Hd;
  signed char* Xq2 = (signed char*)(ws + off); off += (size_t)M * Id;
  signed char* WqU = (signed char*)(ws + off); off += (size_t)Id * Hd;
  signed char* WqD = (signed char*)(ws + off); off += (size_t)Hd * Id;
  float* rs1  = (float*)(ws + off); off += (size_t)M * 4;
  float* rs2  = (float*)(ws + off); off += (size_t)M * 4;
  float* wsum = (float*)(ws + off); off += 256;
  float* Hbuf = (float*)(ws + off); off += (size_t)M * Id * 4;

  const size_t nWup = (size_t)Id * Hd;
  const size_t nWdn = (size_t)Hd * Id;
  const float inv_w = 1.0f / (float)nWup;

  init_kernel<<<1, 32, 0, stream>>>(wsum);
  abssum_kernel<<<1024, 256, 0, stream>>>(wup, wsum + 0, nWup);
  abssum_kernel<<<1024, 256, 0, stream>>>(wdn, wsum + 1, nWdn);
  wquant_kernel<<<2048, 256, 0, stream>>>(wup, WqU, wsum + 0, inv_w, nWup);
  wquant_kernel<<<2048, 256, 0, stream>>>(wdn, WqD, wsum + 1, inv_w, nWdn);

  fwht_quant_kernel<2048><<<M, 256, 0, stream>>>(x, Xq1, rs1);
  gemm_iu8_kernel<<<dim3(Id / 64, M / 128), 256, 0, stream>>>(
      Xq1, WqU, rs1, wsum + 0, inv_w, Hbuf, Id, Hd,  1);

  fwht_quant_kernel<4096><<<M, 256, 0, stream>>>(Hbuf, Xq2, rs2);
  gemm_iu8_kernel<<<dim3(Hd / 64, M / 128), 256, 0, stream>>>(
      Xq2, WqD, rs2, wsum + 1, inv_w, out, Hd, Id,  0);
}
